// Fp8Linear_58463094833530
// MI455X (gfx1250) — 8-bit matrix operand order repaired and checked by reading, compiled, not run on hardware
//
#include <hip/hip_runtime.h>


typedef __attribute__((ext_vector_type(16))) int          v16i;
typedef __attribute__((ext_vector_type(8)))  float        v8f;
typedef __attribute__((ext_vector_type(4)))  float        v4f;
typedef __attribute__((ext_vector_type(4)))  unsigned int v4u;
typedef __attribute__((ext_vector_type(4)))  int          v4i;

#define BM 128
#define BN 128
#define BKK 128
#define LSTRIDE 144
#define TILE_BYTES (128 * LSTRIDE)
#define FP8_MAXV 448.0f

#if __has_builtin(__builtin_amdgcn_global_load_async_to_lds_b128)
#define USE_ASYNC_STAGE 1
typedef __attribute__((address_space(1))) v4i g_v4i;
typedef __attribute__((address_space(3))) v4i l_v4i;
#define ASYNC_CP16(gsrc, ldst)                                                 \
    __builtin_amdgcn_global_load_async_to_lds_b128(                            \
        (g_v4i*)(uintptr_t)(gsrc),                                             \
        (l_v4i*)(unsigned)(uintptr_t)(ldst), 0, 0)
#else
#define USE_ASYNC_STAGE 0
#endif

__device__ __forceinline__ void wait_async_0() {
#if __has_builtin(__builtin_amdgcn_s_wait_asynccnt)
    __builtin_amdgcn_s_wait_asynccnt(0);
#else
    asm volatile("s_wait_asynccnt 0x0" ::: "memory");
#endif
}
__device__ __forceinline__ void wait_async_8() {
#if __has_builtin(__builtin_amdgcn_s_wait_asynccnt)
    __builtin_amdgcn_s_wait_asynccnt(8);
#else
    asm volatile("s_wait_asynccnt 0x8" ::: "memory");
#endif
}

__device__ __forceinline__ float clampq(float v) {
    return fminf(fmaxf(v, -FP8_MAXV), FP8_MAXV);
}

#if !__has_builtin(__builtin_amdgcn_cvt_pk_fp8_f32)
__device__ __forceinline__ unsigned int enc_e4m3_sw(float f) {
    unsigned int u = __float_as_uint(f);
    unsigned int sgn = (u >> 24) & 0x80u;
    float af = fabsf(f);
    if (af < 0.0009765625f) return sgn;
    int ex; float mant = frexpf(af, &ex);
    int E = ex - 1 + 7;
    int m = (int)(mant * 16.0f + 0.5f) - 8;
    if (m == 8) { m = 0; ++E; }
    if (E <= 0) return sgn;
    if (E > 15 || (E == 15 && m > 6)) { E = 15; m = 6; }
    return sgn | ((unsigned int)E << 3) | (unsigned int)m;
}
#endif

__device__ __forceinline__ unsigned int pack4_e4m3(float a, float b, float c, float d) {
#if __has_builtin(__builtin_amdgcn_cvt_pk_fp8_f32)
    int p = __builtin_amdgcn_cvt_pk_fp8_f32(a, b, 0, false);
    p     = __builtin_amdgcn_cvt_pk_fp8_f32(c, d, p, true);
    return (unsigned int)p;
#else
    return enc_e4m3_sw(a) | (enc_e4m3_sw(b) << 8) |
           (enc_e4m3_sw(c) << 16) | (enc_e4m3_sw(d) << 24);
#endif
}

union AFrag { v16i v; unsigned long long q[8]; };
union BFrag { v16i v; v4u q[4]; };

__global__ __launch_bounds__(256)
void quant_x_kernel(const float* __restrict__ x, unsigned char* __restrict__ qa,
                    const float* __restrict__ iscale)
{
    const float inv_s = 1.0f / iscale[0];
    size_t idx = (size_t)blockIdx.x * 256 + threadIdx.x;
    const float* p = x + idx * 16;
    v4u o;
    #pragma unroll
    for (int j = 0; j < 4; ++j) {
        v4f a = *(const v4f*)(p + j * 4);
        o[j] = pack4_e4m3(clampq(a.x * inv_s), clampq(a.y * inv_s),
                          clampq(a.z * inv_s), clampq(a.w * inv_s));
    }
    *(v4u*)(qa + idx * 16) = o;
}

__global__ __launch_bounds__(256)
void quant_w_kernel(const float* __restrict__ qw, unsigned char* __restrict__ qb,
                    int N, int K)
{
    __shared__ unsigned char t[128 * LSTRIDE];
    const int tid = threadIdx.x;
    const int n0 = blockIdx.x * 128;
    const int k0 = blockIdx.y * 128;

    #pragma unroll
    for (int it = 0; it < 16; ++it) {
        int idx = it * 256 + tid;
        int kr  = idx >> 5;
        int c4  = idx & 31;
        v4f w = *(const v4f*)(qw + (size_t)(k0 + kr) * N + n0 + c4 * 4);
        unsigned int pk = pack4_e4m3(clampq(w.x), clampq(w.y),
                                     clampq(w.z), clampq(w.w));
        int nn = c4 * 4;
        t[(nn + 0) * LSTRIDE + kr] = (unsigned char)(pk        & 0xFF);
        t[(nn + 1) * LSTRIDE + kr] = (unsigned char)((pk >> 8)  & 0xFF);
        t[(nn + 2) * LSTRIDE + kr] = (unsigned char)((pk >> 16) & 0xFF);
        t[(nn + 3) * LSTRIDE + kr] = (unsigned char)((pk >> 24) & 0xFF);
    }
    __syncthreads();

    const int n    = tid >> 1;
    const int half = tid & 1;
    #pragma unroll
    for (int j = 0; j < 4; ++j) {
        v4u v = *(const v4u*)(&t[n * LSTRIDE + half * 64 + j * 16]);
        *(v4u*)(qb + (size_t)(n0 + n) * K + k0 + half * 64 + j * 16) = v;
    }
}

__global__ __launch_bounds__(256)
void fp8_gemm_kernel(const unsigned char* __restrict__ qa,
                     const unsigned char* __restrict__ qb,
                     const float* __restrict__ wscale,
                     const float* __restrict__ iscale,
                     const float* __restrict__ qbias,
                     float* __restrict__ out,
                     int M, int N, int K)
{
    extern __shared__ unsigned char smem[];
    const int tid  = threadIdx.x;
    const int lane = tid & 31;
    const int wave = tid >> 5;
    const int wm   = wave >> 2;
    const int wn   = wave & 3;

    const int mBase = blockIdx.y * BM;
    const int nBase = blockIdx.x * BN;

    const unsigned char* gA = qa + (size_t)mBase * K;
    const unsigned char* gB = qb + (size_t)nBase * K;

    v8f acc[4][2];
    #pragma unroll
    for (int i = 0; i < 4; ++i)
        #pragma unroll
        for (int j = 0; j < 2; ++j)
            #pragma unroll
            for (int e = 0; e < 8; ++e)
                acc[i][j][e] = 0.0f;

    const int kTiles = K / BKK;

#if USE_ASYNC_STAGE
    {
        unsigned char* la = smem;
        unsigned char* lb = smem + 2 * TILE_BYTES;
        #pragma unroll
        for (int i = 0; i < 4; ++i) {
            int idx = i * 256 + tid;
            int r = idx >> 3;
            int c = idx & 7;
            ASYNC_CP16(gA + (size_t)r * K + c * 16, la + r * LSTRIDE + c * 16);
            ASYNC_CP16(gB + (size_t)r * K + c * 16, lb + r * LSTRIDE + c * 16);
        }
    }
#else
    v4u sA[4], sB[4];
    #pragma unroll
    for (int i = 0; i < 4; ++i) {
        int idx = i * 256 + tid;
        int r = idx >> 3;
        int c = idx & 7;
        sA[i] = *(const v4u*)(gA + (size_t)r * K + c * 16);
        sB[i] = *(const v4u*)(gB + (size_t)r * K + c * 16);
    }
#endif

    for (int kt = 0; kt < kTiles; ++kt) {
        const int cur = kt & 1;
        unsigned char* laCur = smem + cur * TILE_BYTES;
        unsigned char* lbCur = smem + 2 * TILE_BYTES + cur * TILE_BYTES;

#if USE_ASYNC_STAGE
        if (kt + 1 < kTiles) {
            const int k0n = (kt + 1) * BKK;
            unsigned char* laN = smem + (cur ^ 1) * TILE_BYTES;
            unsigned char* lbN = smem + 2 * TILE_BYTES + (cur ^ 1) * TILE_BYTES;
            #pragma unroll
            for (int i = 0; i < 4; ++i) {
                int idx = i * 256 + tid;
                int r = idx >> 3;
                int c = idx & 7;
                ASYNC_CP16(gA + (size_t)r * K + k0n + c * 16, laN + r * LSTRIDE + c * 16);
                ASYNC_CP16(gB + (size_t)r * K + k0n + c * 16, lbN + r * LSTRIDE + c * 16);
            }
            wait_async_8();
        } else {
            wait_async_0();
        }
        __syncthreads();
#else
        #pragma unroll
        for (int i = 0; i < 4; ++i) {
            int idx = i * 256 + tid;
            int r = idx >> 3;
            int c = idx & 7;
            *(v4u*)(laCur + r * LSTRIDE + c * 16) = sA[i];
            *(v4u*)(lbCur + r * LSTRIDE + c * 16) = sB[i];
        }
        __syncthreads();
        if (kt + 1 < kTiles) {
            const int k0n = (kt + 1) * BKK;
            #pragma unroll
            for (int i = 0; i < 4; ++i) {
                int idx = i * 256 + tid;
                int r = idx >> 3;
                int c = idx & 7;
                sA[i] = *(const v4u*)(gA + (size_t)r * K + k0n + c * 16);
                sB[i] = *(const v4u*)(gB + (size_t)r * K + k0n + c * 16);
            }
        }
#endif

        BFrag bf[2];
        {
            const int koffB = (lane < 16) ? 0 : 16;
            #pragma unroll
            for (int nt = 0; nt < 2; ++nt) {
                int ncol = wn * 32 + nt * 16 + (lane & 15);
                #pragma unroll
                for (int c = 0; c < 4; ++c)
                    bf[nt].q[c] = *(const v4u*)(lbCur + ncol * LSTRIDE + koffB + 32 * c);
            }
        }

        {
            const int koffA = (lane < 16) ? 0 : 16;
            auto loadA = [&](AFrag& f, int mt) {
                int rowT = wm * 64 + mt * 16 + (lane & 15);
                #pragma unroll
                for (int h = 0; h < 2; ++h)
                    #pragma unroll
                    for (int c = 0; c < 4; ++c)
                        f.q[h * 4 + c] = *(const unsigned long long*)
                            (laCur + rowT * LSTRIDE + h * 64 + koffA + (c >> 1) * 32 + (c & 1) * 8);
            };
            AFrag af[2];
            loadA(af[0], 0);
            #pragma unroll
            for (int mt = 0; mt < 4; ++mt) {
                if (mt < 3) loadA(af[(mt + 1) & 1], mt + 1);
                #pragma unroll
                for (int nt = 0; nt < 2; ++nt)
                    acc[mt][nt] = __builtin_amdgcn_wmma_f32_16x16x128_fp8_fp8(
                        af[mt & 1].v, bf[nt].v, (short)0, acc[mt][nt],
                         false,  false);
            }
        }

        __syncthreads();
    }

    const float s_in = iscale[0];
    #pragma unroll
    for (int nt = 0; nt < 2; ++nt) {
        int n = nBase + wn * 32 + nt * 16 + (lane & 15);
        float sc = s_in * wscale[n];
        float bs = qbias[n];
        #pragma unroll
        for (int mt = 0; mt < 4; ++mt) {
            int rbase = mBase + wm * 64 + mt * 16 + ((lane < 16) ? 0 : 8);
            #pragma unroll
            for (int e = 0; e < 8; ++e)
                out[(size_t)(rbase + e) * N + n] = acc[mt][nt][e] * sc + bs;
        }
    }
}

extern "C" void kernel_launch(void* const* d_in, const int* in_sizes, int n_in,
                              void* d_out, int out_size, void* d_ws, size_t ws_size,
                              hipStream_t stream) {
    (void)n_in; (void)out_size; (void)ws_size;
    const float* x      = (const float*)d_in[0];
    const float* qw     = (const float*)d_in[1];
    const float* wscale = (const float*)d_in[2];
    const float* is     = (const float*)d_in[3];
    const float* qb     = (const float*)d_in[4];
    float* out = (float*)d_out;

    const long long N = in_sizes[2];
    const long long K = in_sizes[1] / N;
    const long long M = in_sizes[0] / K;

    unsigned char* qa8 = (unsigned char*)d_ws;
    unsigned char* qb8 = qa8 + (size_t)M * K;

    quant_x_kernel<<<(unsigned)((M * K / 16) / 256), 256, 0, stream>>>(x, qa8, is);
    quant_w_kernel<<<dim3((unsigned)(N / 128), (unsigned)(K / 128)), 256, 0, stream>>>(
        qw, qb8, (int)N, (int)K);
    fp8_gemm_kernel<<<dim3((unsigned)(N / BN), (unsigned)(M / BM)), 256,
                      4 * TILE_BYTES, stream>>>(
        qa8, qb8, wscale, is, qb, out, (int)M, (int)N, (int)K);
}
